// OpinionGAT_3959959847023
// MI455X (gfx1250) — matrix operand order repaired and checked by reading, not compiled, not run on hardware
//
#include <hip/hip_runtime.h>
#include <hip/hip_bf16.h>

// ---------------------------------------------------------------------------
// Types for CDNA5 WMMA
// ---------------------------------------------------------------------------
typedef __attribute__((ext_vector_type(16))) __bf16 v16bf;
typedef __attribute__((ext_vector_type(8)))  float  v8f;

union FragCast {            // 8 dwords <-> 16 bf16 (one WMMA A/B operand, wave32)
    unsigned int u[8];
    v16bf        v;
};

__device__ __forceinline__ unsigned short f2bf(float f) {
    unsigned int u = __float_as_uint(f);
    unsigned int r = u + 0x7FFFu + ((u >> 16) & 1u);   // round-to-nearest-even
    return (unsigned short)(r >> 16);
}

// order-preserving float<->uint encoding for atomicMax on floats
__device__ __forceinline__ unsigned int enc_f(float f) {
    unsigned int u = __float_as_uint(f);
    return (u & 0x80000000u) ? ~u : (u | 0x80000000u);
}
__device__ __forceinline__ float dec_f(unsigned int u) {
    u = (u & 0x80000000u) ? (u & 0x7FFFFFFFu) : ~u;
    return __uint_as_float(u);
}

// edge e -> (src,dst), with implicit self-loops appended after the E0 real edges
__device__ __forceinline__ void get_edge(const long long* __restrict__ ei,
                                         int E0, int e, int& s, int& d) {
    if (e < E0) { s = (int)ei[e]; d = (int)ei[E0 + e]; }
    else        { s = e - E0;     d = e - E0; }
}

// ---------------------------------------------------------------------------
// GEMM: C[M,N] = A[M,K] @ B[K,N] (+bias[N]), fp32 in/out, bf16 WMMA compute.
// Block = 320 threads = 10 waves. Tile: 160(M) x 64(N), K-step 32.
// M % 160 == 0, N % 64 == 0, K % 32 == 0 for all three GEMMs here.
// ---------------------------------------------------------------------------
#define LDSS 40   // padded LDS stride in bf16 elements (20 dwords, conflict-light)

__global__ __launch_bounds__(320)
void gemm_bf16_wmma(const float* __restrict__ A, const float* __restrict__ B,
                    const float* __restrict__ bias, float* __restrict__ C,
                    int M, int K, int Ncols)
{
    __shared__ unsigned short ldsA[160 * LDSS];   // 12.5 KB
    __shared__ unsigned short ldsBT[64 * LDSS];   //  5.0 KB (B stored transposed)

    const int tid  = threadIdx.x;
    const int wave = tid >> 5;
    const int lane = tid & 31;
    const int lo   = lane & 15;
    const int hi   = lane >> 4;

    const int rowBlock = blockIdx.y * 160;
    const int colBlock = blockIdx.x * 64;

    v8f acc[4];
#pragma unroll
    for (int t = 0; t < 4; ++t)
#pragma unroll
        for (int j = 0; j < 8; ++j) acc[t][j] = 0.0f;

    for (int k0 = 0; k0 < K; k0 += 32) {
        // ---- stage A tile (160x32 fp32 -> bf16), 2 threads per row ----
        {
            const int r    = tid >> 1;
            const int half = tid & 1;
            const float* srcp = A + (size_t)(rowBlock + r) * K + k0 + half * 16;
            unsigned short* dstp = ldsA + r * LDSS + half * 16;
#pragma unroll
            for (int i = 0; i < 4; ++i) {
                float4 f = *(const float4*)(srcp + i * 4);
                dstp[i * 4 + 0] = f2bf(f.x);
                dstp[i * 4 + 1] = f2bf(f.y);
                dstp[i * 4 + 2] = f2bf(f.z);
                dstp[i * 4 + 3] = f2bf(f.w);
            }
        }
        // ---- stage B tile (32x64 fp32 -> bf16, transposed into LDS) ----
        for (int idx = tid; idx < (32 * 64) / 4; idx += 320) {
            const int kk = idx >> 4;            // 0..31
            const int cc = (idx & 15) * 4;      // 0..60
            float4 f = *(const float4*)(B + (size_t)(k0 + kk) * Ncols + colBlock + cc);
            ldsBT[(cc + 0) * LDSS + kk] = f2bf(f.x);
            ldsBT[(cc + 1) * LDSS + kk] = f2bf(f.y);
            ldsBT[(cc + 2) * LDSS + kk] = f2bf(f.z);
            ldsBT[(cc + 3) * LDSS + kk] = f2bf(f.w);
        }
        __syncthreads();

        // ---- A fragment (16x32 bf16): row = lane&15, ISA §7.12.2 K mapping ----
        FragCast af;
        {
            const unsigned short* base = ldsA + (wave * 16 + lo) * LDSS;
#pragma unroll
            for (int v = 0; v < 4; ++v) {
                af.u[v]     = *(const unsigned int*)(base + 8 * hi + 2 * v);        // K = 8hi+2v,+1
                af.u[v + 4] = *(const unsigned int*)(base + 16 + 8 * hi + 2 * v);   // K = 16+8hi+2v,+1
            }
        }
        // ---- 4 N-subtiles share the A fragment ----
#pragma unroll
        for (int nt = 0; nt < 4; ++nt) {
            FragCast bf;
            const unsigned short* base = ldsBT + (nt * 16 + lo) * LDSS;   // col=lane&15, K: ISA §7.12.2 mapping (same as A)
#pragma unroll
            for (int p = 0; p < 8; ++p)
                bf.u[p] = *(const unsigned int*)(base + 8 * hi + 2 * (p & 3) + 16 * (p >> 2));
            acc[nt] = __builtin_amdgcn_wmma_f32_16x16x32_bf16(
                false, af.v, false, bf.v, (short)0, acc[nt], false, false);
        }
        __syncthreads();
    }

    // ---- store D (VGPR r: row = r + 8*hi, col = lane&15) ----
#pragma unroll
    for (int nt = 0; nt < 4; ++nt) {
        const int col  = colBlock + nt * 16 + lo;
        const float bv = bias ? bias[col] : 0.0f;
#pragma unroll
        for (int r = 0; r < 8; ++r) {
            const int row = rowBlock + wave * 16 + 8 * hi + r;
            C[(size_t)row * Ncols + col] = acc[nt][r] + bv;
        }
    }
}

// ---------------------------------------------------------------------------
// Per-node attention scalars: alpha_s[n,h] = dot(h[n,h,:], a_src[h,:]), ditto dst.
// One wave per (n,h), C = 128 (4 floats/lane).
// ---------------------------------------------------------------------------
__global__ void alpha_kernel(const float* __restrict__ h,
                             const float* __restrict__ a_s, const float* __restrict__ a_d,
                             float* __restrict__ out_s, float* __restrict__ out_d,
                             int nNodes, int H)
{
    const int gw   = (int)((blockIdx.x * blockDim.x + threadIdx.x) >> 5);
    const int lane = threadIdx.x & 31;
    if (gw >= nNodes * H) return;
    const int n = gw / H, hh = gw % H;

    const float4 hv = *(const float4*)(h + ((size_t)n * H + hh) * 128 + lane * 4);
    const float4 sv = *(const float4*)(a_s + (size_t)hh * 128 + lane * 4);
    const float4 dv = *(const float4*)(a_d + (size_t)hh * 128 + lane * 4);
    float ss = hv.x * sv.x + hv.y * sv.y + hv.z * sv.z + hv.w * sv.w;
    float dd = hv.x * dv.x + hv.y * dv.y + hv.z * dv.z + hv.w * dv.w;
#pragma unroll
    for (int m = 16; m >= 1; m >>= 1) {
        ss += __shfl_xor(ss, m, 32);
        dd += __shfl_xor(dd, m, 32);
    }
    if (lane == 0) { out_s[gw] = ss; out_d[gw] = dd; }
}

// ---------------------------------------------------------------------------
// Edge pass 1: logits + leaky-relu + segment-max (encoded-uint atomicMax)
// ---------------------------------------------------------------------------
__global__ void edge_max_kernel(const long long* __restrict__ ei, int E0, int E,
                                const float* __restrict__ as_, const float* __restrict__ ad_,
                                unsigned int* __restrict__ m_enc, float* __restrict__ ebuf, int H)
{
    const int i = blockIdx.x * blockDim.x + threadIdx.x;
    if (i >= E * H) return;
    const int e = i / H, hh = i % H;
    int s, d; get_edge(ei, E0, e, s, d);
    float v = as_[s * H + hh] + ad_[d * H + hh];
    v = (v > 0.0f) ? v : 0.2f * v;            // leaky_relu(0.2)
    ebuf[i] = v;
    atomicMax(&m_enc[d * H + hh], enc_f(v));
}

// ---------------------------------------------------------------------------
// Edge pass 2: exp(e - max) + segment-sum
// ---------------------------------------------------------------------------
__global__ void edge_expsum_kernel(const long long* __restrict__ ei, int E0, int E,
                                   const unsigned int* __restrict__ m_enc,
                                   float* __restrict__ z, float* __restrict__ ebuf, int H)
{
    const int i = blockIdx.x * blockDim.x + threadIdx.x;
    if (i >= E * H) return;
    const int e = i / H, hh = i % H;
    int s, d; get_edge(ei, E0, e, s, d);
    const float m = dec_f(m_enc[d * H + hh]);
    const float a = __expf(ebuf[i] - m);
    ebuf[i] = a;
    atomicAdd(&z[d * H + hh], a);
}

// ---------------------------------------------------------------------------
// Edge pass 3: out[dst,h,:] += (a/z) * h[src,h,:]   (wave per (e,h), C = 128)
// ---------------------------------------------------------------------------
__global__ void edge_agg_kernel(const long long* __restrict__ ei, int E0, int E,
                                const float* __restrict__ ebuf, const float* __restrict__ z,
                                const float* __restrict__ h, float* __restrict__ agg, int H)
{
    const int gw   = (int)((blockIdx.x * blockDim.x + threadIdx.x) >> 5);
    const int lane = threadIdx.x & 31;
    if (gw >= E * H) return;
    const int e = gw / H, hh = gw % H;
    int s, d; get_edge(ei, E0, e, s, d);
    const float w = ebuf[gw] / (z[d * H + hh] + 1e-16f);
    const float4 hv = *(const float4*)(h + ((size_t)s * H + hh) * 128 + lane * 4);
    float* dp = agg + ((size_t)d * H + hh) * 128 + lane * 4;
    atomicAdd(dp + 0, w * hv.x);
    atomicAdd(dp + 1, w * hv.y);
    atomicAdd(dp + 2, w * hv.z);
    atomicAdd(dp + 3, w * hv.w);
}

// ---------------------------------------------------------------------------
// bias add (+ optional ELU), in place
// ---------------------------------------------------------------------------
__global__ void bias_act_kernel(float* __restrict__ io, const float* __restrict__ bias,
                                long long total, int F, int do_elu)
{
    const long long i = (long long)blockIdx.x * blockDim.x + threadIdx.x;
    if (i >= total) return;
    float v = io[i] + bias[(int)(i % F)];
    if (do_elu) v = (v > 0.0f) ? v : (__expf(v) - 1.0f);
    io[i] = v;
}

__global__ void fill_f32(float* __restrict__ p, float v, long long n) {
    const long long i = (long long)blockIdx.x * blockDim.x + threadIdx.x;
    if (i < n) p[i] = v;
}
__global__ void fill_u32(unsigned int* __restrict__ p, unsigned int v, long long n) {
    const long long i = (long long)blockIdx.x * blockDim.x + threadIdx.x;
    if (i < n) p[i] = v;
}

// ---------------------------------------------------------------------------
extern "C" void kernel_launch(void* const* d_in, const int* in_sizes, int n_in,
                              void* d_out, int out_size, void* d_ws, size_t ws_size,
                              hipStream_t stream)
{
    const float*     x   = (const float*)d_in[0];
    const long long* ei  = (const long long*)d_in[1];   // [2, 320000] int64
    const float*     Wp  = (const float*)d_in[2];
    const float*     bp  = (const float*)d_in[3];
    const float*     W1  = (const float*)d_in[4];
    const float*     as1 = (const float*)d_in[5];
    const float*     ad1 = (const float*)d_in[6];
    const float*     b1  = (const float*)d_in[7];
    const float*     W2  = (const float*)d_in[8];
    const float*     as2 = (const float*)d_in[9];
    const float*     ad2 = (const float*)d_in[10];
    const float*     b2  = (const float*)d_in[11];
    float*           out = (float*)d_out;

    const int Nn = 20000, E0 = 320000, E = 340000;

    // workspace layout (floats)
    float* ws = (float*)d_ws;
    size_t off = 0;
    float*        h0   = ws + off; off += (size_t)Nn * 128;   // also reused as h2
    float*        h1   = ws + off; off += (size_t)Nn * 512;
    float*        agg1 = ws + off; off += (size_t)Nn * 512;   // becomes h1act in place
    float*        aS1  = ws + off; off += (size_t)Nn * 4;
    float*        aD1  = ws + off; off += (size_t)Nn * 4;
    unsigned int* m1   = (unsigned int*)(ws + off); off += (size_t)Nn * 4;
    float*        z1   = ws + off; off += (size_t)Nn * 4;
    float*        e1   = ws + off; off += (size_t)E * 4;
    float*        aS2  = ws + off; off += (size_t)Nn;
    float*        aD2  = ws + off; off += (size_t)Nn;
    unsigned int* m2   = (unsigned int*)(ws + off); off += (size_t)Nn;
    float*        z2   = ws + off; off += (size_t)Nn;
    float*        e2   = ws + off; off += (size_t)E;
    (void)ws_size; (void)in_sizes; (void)n_in; (void)out_size;

    const int T = 256;
    auto blk = [](long long n, int t) { return (unsigned)((n + t - 1) / t); };

    // ---- projection: h0 = x @ Wp + bp  (20000x768 @ 768x128) ----
    gemm_bf16_wmma<<<dim3(128 / 64, Nn / 160), 320, 0, stream>>>(x, Wp, bp, h0, Nn, 768, 128);

    // ---- layer 1 (H=4, C=128) ----
    gemm_bf16_wmma<<<dim3(512 / 64, Nn / 160), 320, 0, stream>>>(h0, W1, nullptr, h1, Nn, 128, 512);
    alpha_kernel<<<blk((long long)Nn * 4, 8), 256, 0, stream>>>(h1, as1, ad1, aS1, aD1, Nn, 4);
    fill_u32<<<blk((long long)Nn * 4, T), T, 0, stream>>>(m1, 0u, (long long)Nn * 4);
    fill_f32<<<blk((long long)Nn * 4, T), T, 0, stream>>>(z1, 0.0f, (long long)Nn * 4);
    fill_f32<<<blk((long long)Nn * 512, T), T, 0, stream>>>(agg1, 0.0f, (long long)Nn * 512);
    edge_max_kernel   <<<blk((long long)E * 4, T), T, 0, stream>>>(ei, E0, E, aS1, aD1, m1, e1, 4);
    edge_expsum_kernel<<<blk((long long)E * 4, T), T, 0, stream>>>(ei, E0, E, m1, z1, e1, 4);
    edge_agg_kernel   <<<blk((long long)E * 4, 8), 256, 0, stream>>>(ei, E0, E, e1, z1, h1, agg1, 4);
    bias_act_kernel   <<<blk((long long)Nn * 512, T), T, 0, stream>>>(agg1, b1, (long long)Nn * 512, 512, 1);

    // ---- layer 2 (H=1, C=128): h2 = elu(h1agg) @ W2 (reuse h0 buffer) ----
    gemm_bf16_wmma<<<dim3(128 / 64, Nn / 160), 320, 0, stream>>>(agg1, W2, nullptr, h0, Nn, 512, 128);
    alpha_kernel<<<blk((long long)Nn, 8), 256, 0, stream>>>(h0, as2, ad2, aS2, aD2, Nn, 1);
    fill_u32<<<blk((long long)Nn, T), T, 0, stream>>>(m2, 0u, (long long)Nn);
    fill_f32<<<blk((long long)Nn, T), T, 0, stream>>>(z2, 0.0f, (long long)Nn);
    fill_f32<<<blk((long long)Nn * 128, T), T, 0, stream>>>(out, 0.0f, (long long)Nn * 128);
    edge_max_kernel   <<<blk((long long)E, T), T, 0, stream>>>(ei, E0, E, aS2, aD2, m2, e2, 1);
    edge_expsum_kernel<<<blk((long long)E, T), T, 0, stream>>>(ei, E0, E, m2, z2, e2, 1);
    edge_agg_kernel   <<<blk((long long)E, 8), 256, 0, stream>>>(ei, E0, E, e2, z2, h0, out, 1);
    bias_act_kernel   <<<blk((long long)Nn * 128, T), T, 0, stream>>>(out, b2, (long long)Nn * 128, 128, 0);
}
